// SpecialTimedDecoderBlock_32847909880277
// MI455X (gfx1250) — matrix operand order repaired and checked by reading, not compiled, not run on hardware
//
#include <hip/hip_runtime.h>
#include <hip/hip_bf16.h>
#include <math.h>

#define NH   16
#define CTX  1024
#define DM   1024
#define HDIM 64
#define LN_EPS 1e-5f

typedef __attribute__((ext_vector_type(16))) __bf16 v16bf;
typedef __attribute__((ext_vector_type(4)))  __bf16 v4bf;
typedef __attribute__((ext_vector_type(8)))  float  v8f;

union Frag16 { v16bf v; unsigned u[8]; };

#if __has_builtin(__builtin_amdgcn_global_load_async_to_lds_b128) && \
    __has_builtin(__builtin_amdgcn_s_wait_asynccnt)
#define USE_ASYNC 1
typedef int v4i __attribute__((vector_size(16)));
typedef __attribute__((address_space(1))) v4i v4i_g;
typedef __attribute__((address_space(3))) v4i v4i_l;
#else
#define USE_ASYNC 0
#endif

__device__ __forceinline__ __bf16 f2bf(float f) {
  unsigned u = __builtin_bit_cast(unsigned, f);
  u += 0x7FFFu + ((u >> 16) & 1u);  // round-to-nearest-even
  unsigned short s = (unsigned short)(u >> 16);
  return __builtin_bit_cast(__bf16, s);
}

// 16-byte global -> LDS copy; async DMA (ASYNCcnt) when the toolchain has it.
__device__ __forceinline__ void cp16(const __bf16* g, __bf16* l) {
#if USE_ASYNC
  __builtin_amdgcn_global_load_async_to_lds_b128((v4i_g*)g, (v4i_l*)l, 0, 0);
#else
  *(uint4*)l = *(const uint4*)g;
#endif
}
__device__ __forceinline__ void cp_wait() {
#if USE_ASYNC
  __builtin_amdgcn_s_wait_asynccnt(0);
#endif
}

// A-matrix 16x32 bf16 fragment (ISA 7.12.2): lane&15 = row M,
// VGPR0..3 -> K = 8*(lane>>4)+{0..7}, VGPR4..7 -> K = 16+8*(lane>>4)+{0..7}
__device__ __forceinline__ v16bf load_frag_a(const __bf16* base, int ld, int lane) {
  int r  = lane & 15;
  int ko = (lane >> 4) << 3;
  const __bf16* p = base + r * ld + ko;
  Frag16 f;
#pragma unroll
  for (int i = 0; i < 4; ++i) {
    f.u[i]     = *(const unsigned*)(p + 2 * i);
    f.u[i + 4] = *(const unsigned*)(p + 16 + 2 * i);
  }
  return f.v;
}

// B-matrix 32x16 bf16 fragment: lane&15 = col N, VGPR0..3 -> K = 8*(lane>>4)+{0..7}, VGPR4..7 -> K = 16+8*(lane>>4)+{0..7}.
// LDS holds B as [n][k].
__device__ __forceinline__ v16bf load_frag_b(const __bf16* base, int ld, int lane) {
  int n  = lane & 15;
  int ko = (lane >> 4) << 3;
  const __bf16* p = base + n * ld + ko;
  Frag16 f;
#pragma unroll
  for (int i = 0; i < 4; ++i) {
    f.u[i]     = *(const unsigned*)(p + 2 * i);
    f.u[i + 4] = *(const unsigned*)(p + 16 + 2 * i);
  }
  return f.v;
}

__device__ __forceinline__ v8f wmma_bf16(v16bf a, v16bf b, v8f c) {
  return __builtin_amdgcn_wmma_f32_16x16x32_bf16(false, a, false, b, (short)0, c,
                                                 false, false);
}

__device__ __forceinline__ float rowmax16(float v) {
#pragma unroll
  for (int m = 8; m >= 1; m >>= 1) v = fmaxf(v, __shfl_xor(v, m, 16));
  return v;
}
__device__ __forceinline__ float rowsum16(float v) {
#pragma unroll
  for (int m = 8; m >= 1; m >>= 1) v += __shfl_xor(v, m, 16);
  return v;
}

// ---------------------------------------------------------------------------
// One-time weight convert+transpose: Wt[n*K+k] = bf16(W[k*ldb + (n>>6)*colBlock
// + (n&63)]).  Plain row-major: ldb=N, colBlock=64. Per-head (NH,DM,HDIM):
// ldb=HDIM, colBlock=DM*HDIM.
// ---------------------------------------------------------------------------
__global__ void __launch_bounds__(256)
wconv_kernel(const float* __restrict__ W, __bf16* __restrict__ Wt,
             int K, int N, int ldb, int colBlock) {
  int idx = blockIdx.x * 256 + threadIdx.x;
  if (idx >= N * K) return;
  int n = idx / K, k = idx - n * K;
  Wt[idx] = f2bf(W[(size_t)k * ldb + (size_t)(n >> 6) * colBlock + (n & 63)]);
}

// ---------------------------------------------------------------------------
// bf16 WMMA GEMM, BM=128 BN=64 BK=64, double-buffered LDS, async tile DMA.
// A: bf16, plain [M][K] or (AHEADED) [B,NH,T,HDIM]. Bt: bf16 [N][K].
// outMode: 0 = f32 plain (+bias/+gelu/+resid)  1 = bf16 headed [B,NH,T,HDIM]
//          2 = bf16 headedT [B,NH,HDIM,T]      3 = bf16 plain [M][N]
// ---------------------------------------------------------------------------
#define BM 128
#define BN 64
#define BK 64
#define LDT 80  // LDS row stride (bf16 elems): 160B, 16B multiple

template <int AHEADED>
__global__ void __launch_bounds__(256)
gemm_kernel(const __bf16* __restrict__ A, const __bf16* __restrict__ Bt,
            const float* __restrict__ bias, const float* __restrict__ resid,
            void* __restrict__ Cout, int M, int N, int K,
            int outMode, int doGelu) {
  __shared__ __attribute__((aligned(16))) __bf16 sA[2][BM][LDT];
  __shared__ __attribute__((aligned(16))) __bf16 sB[2][BN][LDT];

  const int tid = threadIdx.x;
  const int lane = tid & 31, wave = tid >> 5;
  const int wm = wave >> 1, wn = wave & 1;
  const int mBase = blockIdx.y * BM;
  const int nBase = blockIdx.x * BN;

  v8f acc[2][2] = {};

  auto loadStage = [&](int k0, int st) {
    // A tile: 128 rows x 8 chunks of 16B
#pragma unroll
    for (int ii = 0; ii < 4; ++ii) {
      int i = tid + ii * 256;
      int r = i >> 3, c = (i & 7) << 3;
      const __bf16* g;
      if (AHEADED) {
        int m = mBase + r, k = k0 + c;
        int bb = m >> 10, t = m & 1023, head = k >> 6, hd = k & 63;
        g = A + (size_t)((((bb << 4) + head) << 10) + t) * HDIM + hd;
      } else {
        g = A + (size_t)(mBase + r) * K + k0 + c;
      }
      cp16(g, &sA[st][r][c]);
    }
    // B tile: 64 rows x 8 chunks
#pragma unroll
    for (int ii = 0; ii < 2; ++ii) {
      int i = tid + ii * 256;
      int n = i >> 3, c = (i & 7) << 3;
      cp16(Bt + (size_t)(nBase + n) * K + k0 + c, &sB[st][n][c]);
    }
  };

  loadStage(0, 0);
  cp_wait();
  __syncthreads();

  const int NT = K / BK;
  for (int kt = 0; kt < NT; ++kt) {
    const int cur = kt & 1;
    if (kt + 1 < NT) loadStage((kt + 1) * BK, cur ^ 1);  // DMA overlaps WMMA
#pragma unroll
    for (int ks = 0; ks < 2; ++ks) {
      v16bf a0 = load_frag_a(&sA[cur][wm * 32][ks * 32],      LDT, lane);
      v16bf a1 = load_frag_a(&sA[cur][wm * 32 + 16][ks * 32], LDT, lane);
      v16bf b0 = load_frag_b(&sB[cur][wn * 32][ks * 32],      LDT, lane);
      v16bf b1 = load_frag_b(&sB[cur][wn * 32 + 16][ks * 32], LDT, lane);
      acc[0][0] = wmma_bf16(a0, b0, acc[0][0]);
      acc[0][1] = wmma_bf16(a0, b1, acc[0][1]);
      acc[1][0] = wmma_bf16(a1, b0, acc[1][0]);
      acc[1][1] = wmma_bf16(a1, b1, acc[1][1]);
    }
    cp_wait();
    __syncthreads();
  }

  const int rbase = (lane >> 4) << 3;
  const int cL = lane & 15;
#pragma unroll
  for (int mf = 0; mf < 2; ++mf)
#pragma unroll
    for (int nf = 0; nf < 2; ++nf)
#pragma unroll
      for (int j = 0; j < 8; ++j) {
        int row = mBase + wm * 32 + mf * 16 + rbase + j;
        int col = nBase + wn * 32 + nf * 16 + cL;
        float v = acc[mf][nf][j];
        if (bias) v += bias[col];
        if (doGelu) v = 0.5f * v * (1.0f + erff(v * 0.70710678f));
        if (outMode == 0) {
          float* C = (float*)Cout;
          if (resid) v += resid[(size_t)row * N + col];
          C[(size_t)row * N + col] = v;
        } else {
          __bf16* C = (__bf16*)Cout;
          size_t oidx;
          int bb = row >> 10, t = row & 1023, head = col >> 6, hd = col & 63;
          if (outMode == 1)
            oidx = (size_t)((((bb << 4) + head) << 10) + t) * HDIM + hd;
          else if (outMode == 2)
            oidx = (size_t)((((bb << 4) + head) << 6) + hd) * CTX + t;
          else
            oidx = (size_t)row * N + col;
          C[oidx] = f2bf(v);
        }
      }
}

// ---------------------------------------------------------------------------
// Flash attention: Q,K bf16 [B*NH][CTX][HDIM], Vt bf16 [B*NH][HDIM][CTX].
// Causal, scale = 1/32. 8 waves x 32 query rows; 32-key tiles via async DMA.
// Output att bf16 [B*NH][CTX][HDIM].
// ---------------------------------------------------------------------------
__global__ void __launch_bounds__(256)
attn_kernel(const __bf16* __restrict__ Q, const __bf16* __restrict__ Kp,
            const __bf16* __restrict__ Vt, __bf16* __restrict__ O) {
  __shared__ __attribute__((aligned(16))) __bf16 sQ[256][64];
  __shared__ __attribute__((aligned(16))) __bf16 sK[32][80];
  __shared__ __attribute__((aligned(16))) __bf16 sVt[64][40];
  __shared__ __attribute__((aligned(16))) __bf16 sP[8][32][40];

  const int bh = blockIdx.x;
  const int q0 = blockIdx.y * 256;
  const int tid = threadIdx.x, lane = tid & 31, wave = tid >> 5;
  const int rbase = (lane >> 4) << 3;
  const int cL = lane & 15;
  const float SCALE = 0.03125f;  // 1024^-0.5

  const __bf16* Qb = Q  + (size_t)bh * CTX * HDIM;
  const __bf16* Kb = Kp + (size_t)bh * CTX * HDIM;
  const __bf16* Vb = Vt + (size_t)bh * HDIM * CTX;

  // stage Q block (256x64) via 16B copies
#pragma unroll
  for (int ii = 0; ii < 8; ++ii) {
    int i = tid + ii * 256;
    int r = i >> 3, c = (i & 7) << 3;
    cp16(Qb + (size_t)(q0 + r) * HDIM + c, &sQ[r][c]);
  }
  cp_wait();
  __syncthreads();

  v16bf qf[2][2];
#pragma unroll
  for (int mf = 0; mf < 2; ++mf)
#pragma unroll
    for (int ks = 0; ks < 2; ++ks)
      qf[mf][ks] = load_frag_a(&sQ[wave * 32 + mf * 16][ks * 32], 64, lane);

  v8f o[2][4] = {};
  float mst[2][8], lst[2][8];
#pragma unroll
  for (int mf = 0; mf < 2; ++mf)
#pragma unroll
    for (int j = 0; j < 8; ++j) { mst[mf][j] = -1e30f; lst[mf][j] = 0.0f; }

  const int kend = q0 + 256;  // causal bound (uniform per block)
  for (int kb = 0; kb < kend; kb += 32) {
    {  // K tile 32x64 (8 chunks/row) + V tile 64x32 (4 chunks/row)
      int r = tid >> 3, c = (tid & 7) << 3;
      cp16(Kb + (size_t)(kb + r) * HDIM + c, &sK[r][c]);
      int hd = tid >> 2, c2 = (tid & 3) << 3;
      cp16(Vb + (size_t)hd * CTX + kb + c2, &sVt[hd][c2]);
    }
    cp_wait();
    __syncthreads();

    // S = Q K^T (32x32 per wave)
    v8f s[2][2];
#pragma unroll
    for (int mf = 0; mf < 2; ++mf)
#pragma unroll
      for (int nf = 0; nf < 2; ++nf) {
        v8f t = {};
        t = wmma_bf16(qf[mf][0], load_frag_b(&sK[nf * 16][0],  80, lane), t);
        t = wmma_bf16(qf[mf][1], load_frag_b(&sK[nf * 16][32], 80, lane), t);
        s[mf][nf] = t;
      }

    // online softmax (row = mf*16 + rbase + j, col = nf*16 + cL)
#pragma unroll
    for (int mf = 0; mf < 2; ++mf) {
      float sv0[8], sv1[8];
#pragma unroll
      for (int j = 0; j < 8; ++j) {
        int qi = q0 + wave * 32 + mf * 16 + rbase + j;
        float a0 = s[mf][0][j] * SCALE;
        float a1 = s[mf][1][j] * SCALE;
        sv0[j] = (kb + cL <= qi)      ? a0 : -1e30f;
        sv1[j] = (kb + 16 + cL <= qi) ? a1 : -1e30f;
      }
#pragma unroll
      for (int j = 0; j < 8; ++j) {
        float rm = rowmax16(fmaxf(sv0[j], sv1[j]));
        float mn = fmaxf(mst[mf][j], rm);
        float al = __expf(mst[mf][j] - mn);
        float p0 = __expf(sv0[j] - mn);
        float p1 = __expf(sv1[j] - mn);
        float rs = rowsum16(p0 + p1);
        lst[mf][j] = lst[mf][j] * al + rs;
        mst[mf][j] = mn;
#pragma unroll
        for (int vf = 0; vf < 4; ++vf) o[mf][vf][j] *= al;
        int prow = mf * 16 + rbase + j;
        sP[wave][prow][cL]      = f2bf(p0);
        sP[wave][prow][16 + cL] = f2bf(p1);
      }
    }
    __syncthreads();  // P staged; also fences sK readers

    // O += P @ V
#pragma unroll
    for (int mf = 0; mf < 2; ++mf) {
      v16bf pf = load_frag_a(&sP[wave][mf * 16][0], 40, lane);
#pragma unroll
      for (int vf = 0; vf < 4; ++vf)
        o[mf][vf] = wmma_bf16(pf, load_frag_b(&sVt[vf * 16][0], 40, lane), o[mf][vf]);
    }
    __syncthreads();  // protect sK/sVt/sP before next tile
  }

  __bf16* Ob = O + (size_t)bh * CTX * HDIM;
#pragma unroll
  for (int mf = 0; mf < 2; ++mf)
#pragma unroll
    for (int j = 0; j < 8; ++j) {
      float inv = 1.0f / lst[mf][j];
      int row = q0 + wave * 32 + mf * 16 + rbase + j;
#pragma unroll
      for (int vf = 0; vf < 4; ++vf)
        Ob[(size_t)row * HDIM + vf * 16 + cL] = f2bf(o[mf][vf][j] * inv);
    }
}

// ---------------------------------------------------------------------------
// LayerNorm over last dim (1024), one row per block, bf16 output.
// ---------------------------------------------------------------------------
__global__ void __launch_bounds__(256)
ln_kernel(const float* __restrict__ x, const float* __restrict__ w,
          const float* __restrict__ b, __bf16* __restrict__ y) {
  __shared__ float rs[256], rs2[256];
  const int row = blockIdx.x, tid = threadIdx.x;
  const float4* xr = (const float4*)(x + (size_t)row * DM);
  float4 v = xr[tid];
  float s  = v.x + v.y + v.z + v.w;
  float s2 = v.x * v.x + v.y * v.y + v.z * v.z + v.w * v.w;
  rs[tid] = s; rs2[tid] = s2;
  __syncthreads();
  for (int off = 128; off > 0; off >>= 1) {
    if (tid < off) { rs[tid] += rs[tid + off]; rs2[tid] += rs2[tid + off]; }
    __syncthreads();
  }
  float mean = rs[0] * (1.0f / DM);
  float var  = rs2[0] * (1.0f / DM) - mean * mean;
  float inv  = rsqrtf(var + LN_EPS);
  float4 wv = ((const float4*)w)[tid];
  float4 bv = ((const float4*)b)[tid];
  v4bf out;
  out[0] = f2bf((v.x - mean) * inv * wv.x + bv.x);
  out[1] = f2bf((v.y - mean) * inv * wv.y + bv.y);
  out[2] = f2bf((v.z - mean) * inv * wv.z + bv.z);
  out[3] = f2bf((v.w - mean) * inv * wv.w + bv.w);
  *(v4bf*)(y + (size_t)row * DM + tid * 4) = out;
}

// x = cur + new + emb_row  (float4)
__global__ void __launch_bounds__(256)
add_kernel(const float4* __restrict__ a, const float4* __restrict__ b,
           const float4* __restrict__ e, float4* __restrict__ o, int n4) {
  int i = blockIdx.x * 256 + threadIdx.x;
  if (i >= n4) return;
  float4 av = a[i], bv = b[i], ev = e[i & (DM / 4 - 1)];
  float4 r;
  r.x = av.x + bv.x + ev.x; r.y = av.y + bv.y + ev.y;
  r.z = av.z + bv.z + ev.z; r.w = av.w + bv.w + ev.w;
  o[i] = r;
}

// ---------------------------------------------------------------------------
extern "C" void kernel_launch(void* const* d_in, const int* in_sizes, int n_in,
                              void* d_out, int out_size, void* d_ws, size_t ws_size,
                              hipStream_t stream) {
  (void)in_sizes; (void)n_in; (void)out_size; (void)ws_size;
  const float* x_in = (const float*)d_in[0];
  const float* emb  = (const float*)d_in[1];
  const float* Wq   = (const float*)d_in[2];
  const float* Wk   = (const float*)d_in[3];
  const float* Wv   = (const float*)d_in[4];
  const float* fc_w = (const float*)d_in[5];
  const float* fc_b = (const float*)d_in[6];
  const float* ln1w = (const float*)d_in[7];
  const float* ln1b = (const float*)d_in[8];
  const float* ln2w = (const float*)d_in[9];
  const float* ln2b = (const float*)d_in[10];
  const float* w1   = (const float*)d_in[11];
  const float* b1   = (const float*)d_in[12];
  const float* w2   = (const float*)d_in[13];
  const float* b2   = (const float*)d_in[14];
  const float* w3   = (const float*)d_in[15];
  const float* b3   = (const float*)d_in[16];

  const size_t SZ = (size_t)4 * CTX * DM;  // 4M elems
  char* p = (char*)d_ws;
  auto carve = [&](size_t bytes) {
    char* r = p;
    p += (bytes + 255) & ~(size_t)255;
    return r;
  };
  float*  x    = (float*)carve(SZ * 4);
  float*  nb   = (float*)carve(SZ * 4);
  __bf16* h    = (__bf16*)carve(SZ * 2);
  __bf16* h2   = (__bf16*)carve(SZ * 2);
  __bf16* qb   = (__bf16*)carve(SZ * 2);
  __bf16* kbuf = (__bf16*)carve(SZ * 2);
  __bf16* vt   = (__bf16*)carve(SZ * 2);
  __bf16* att  = (__bf16*)carve(SZ * 2);
  __bf16* m1   = (__bf16*)carve(SZ * 4);  // 4096x2048
  __bf16* m2   = (__bf16*)carve(SZ * 2);
  __bf16* WqT  = (__bf16*)carve((size_t)DM * DM * 2);
  __bf16* WkT  = (__bf16*)carve((size_t)DM * DM * 2);
  __bf16* WvT  = (__bf16*)carve((size_t)DM * DM * 2);
  __bf16* fcT  = (__bf16*)carve((size_t)DM * DM * 2);
  __bf16* w1T  = (__bf16*)carve((size_t)DM * 2 * DM * 2);
  __bf16* w2T  = (__bf16*)carve((size_t)DM * 2 * DM * 2);
  __bf16* w3T  = (__bf16*)carve((size_t)DM * DM * 2);

  const int M = 4 * CTX;  // 4096

  // One-time weight convert + transpose to [N][K] bf16
  {
    int g1 = (DM * DM) / 256, g2 = (2 * DM * DM) / 256;
    wconv_kernel<<<g1, 256, 0, stream>>>(Wq, WqT, DM, DM, HDIM, DM * HDIM);
    wconv_kernel<<<g1, 256, 0, stream>>>(Wk, WkT, DM, DM, HDIM, DM * HDIM);
    wconv_kernel<<<g1, 256, 0, stream>>>(Wv, WvT, DM, DM, HDIM, DM * HDIM);
    wconv_kernel<<<g1, 256, 0, stream>>>(fc_w, fcT, DM, DM, DM, 64);
    wconv_kernel<<<g2, 256, 0, stream>>>(w1, w1T, DM, 2 * DM, 2 * DM, 64);
    wconv_kernel<<<g2, 256, 0, stream>>>(w2, w2T, 2 * DM, DM, DM, 64);
    wconv_kernel<<<g1, 256, 0, stream>>>(w3, w3T, DM, DM, DM, 64);
  }

  auto gemmP = [&](const __bf16* A, const __bf16* Bt, const float* bias,
                   const float* resid, void* C, int N, int K, int outMode,
                   int doGelu) {
    dim3 g(N / BN, M / BM);
    gemm_kernel<0><<<g, 256, 0, stream>>>(A, Bt, bias, resid, C, M, N, K,
                                          outMode, doGelu);
  };
  auto gemmH = [&](const __bf16* A, const __bf16* Bt, const float* bias,
                   const float* resid, void* C, int N, int K, int outMode,
                   int doGelu) {
    dim3 g(N / BN, M / BM);
    gemm_kernel<1><<<g, 256, 0, stream>>>(A, Bt, bias, resid, C, M, N, K,
                                          outMode, doGelu);
  };

  for (int t = 0; t < 8; ++t) {
    const float* curS = (t == 0) ? x_in : x;
    const float* newS = (t == 0) ? x_in : nb;
    add_kernel<<<(int)(SZ / 4 / 256), 256, 0, stream>>>(
        (const float4*)curS, (const float4*)newS,
        (const float4*)(emb + t * DM), (float4*)x, (int)(SZ / 4));
    ln_kernel<<<M, 256, 0, stream>>>(x, ln1w, ln1b, h);

    gemmP(h, WqT, nullptr, nullptr, qb,   DM, DM, 1, 0);  // q [B,H,T,HD]
    gemmP(h, WkT, nullptr, nullptr, kbuf, DM, DM, 1, 0);  // k [B,H,T,HD]
    gemmP(h, WvT, nullptr, nullptr, vt,   DM, DM, 2, 0);  // v [B,H,HD,T]

    attn_kernel<<<dim3(4 * NH, CTX / 256), 256, 0, stream>>>(qb, kbuf, vt, att);

    // fc: headed bf16 A, f32 out with residual (in-place x update)
    gemmH(att, fcT, fc_b, x, x, DM, DM, 0, 0);

    ln_kernel<<<M, 256, 0, stream>>>(x, ln2w, ln2b, h2);

    gemmP(h2, w1T, b1, nullptr, m1, 2 * DM, DM, 3, 0);
    gemmP(m1, w2T, b2, nullptr, m2, DM, 2 * DM, 3, 0);

    float* outp = (t == 7) ? (float*)d_out : nb;
    gemmP(m2, w3T, b3, x, outp, DM, DM, 0, 1);  // gelu then +x
  }
}
